// Solograph_79456894976244
// MI455X (gfx1250) — matrix operand order repaired and checked by reading, not compiled, not run on hardware
//
#include <hip/hip_runtime.h>

// ---------------------------------------------------------------------------
// Problem constants (match reference)
// ---------------------------------------------------------------------------
#define NB   64            // B scenes
#define PP   512           // nodes per scene per modality
#define NN   (NB*PP)       // 32768 nodes per modality
#define NZ_  64
#define DD   256
#define HH   4
#define DH_  64
#define LL   2
#define RR   13
#define ECROSS 131072

typedef __attribute__((ext_vector_type(16))) __bf16 v16bf;
typedef __attribute__((ext_vector_type(2)))  __bf16 v2bf;
typedef __attribute__((ext_vector_type(2)))  float  v2f;
typedef __attribute__((ext_vector_type(8)))  float  v8f;

// ---------------------------------------------------------------------------
// gfx1250 async global->LDS copy (ASYNCcnt-tracked, CDNA5 ISA)
// ---------------------------------------------------------------------------
static __device__ __forceinline__ void async_load_b128(unsigned lds_off,
                                                       const float* gbase,
                                                       unsigned byte_off) {
  asm volatile("global_load_async_to_lds_b128 %0, %1, %2"
               :: "v"(lds_off), "v"(byte_off), "s"(gbase) : "memory");
}

// ---------------------------------------------------------------------------
// One-time weight pre-pass: f32 [K,N] row-major -> bf16 transposed [N,K].
// Puts each WMMA B-fragment (16 consecutive K for one column) on a single
// aligned 32B global load; weights stay resident in the 192MB L2.
// ---------------------------------------------------------------------------
__global__ void k_wswz(const float* __restrict__ Wsrc, __bf16* __restrict__ Wdst,
                       int nmat) {
  int t = blockIdx.x * blockDim.x + threadIdx.x;
  if (t >= nmat * DD * DD) return;
  int m = t >> 16;
  int r = t & 0xFFFF;
  int n = r >> 8, k = r & 255;
  Wdst[(size_t)m * DD * DD + n * DD + k] = (__bf16)Wsrc[(size_t)m * DD * DD + k * DD + n];
}

// ---------------------------------------------------------------------------
// WMMA GEMM:  C[M,256] = A[M,256] @ W[256,256] + bias, optional fused ReLU.
// Block tile 128x64, 8 waves (wave32), each wave owns a 32x32 patch as
// 2x2 v_wmma_f32_16x16x32_bf16 tiles; K fully unrolled (8 steps).
// A: f32 async-staged to LDS, double-buffered on ASYNCcnt in-order completion,
//    packed-converted to bf16 in-register. B: global loads of pre-swizzled bf16.
// ---------------------------------------------------------------------------
#define TM 128
#define TN 64
#define TK 32

union AFrag { v16bf v; v2bf h[8]; };

__global__ __launch_bounds__(256) void k_gemm256(
    const float* __restrict__ A, const __bf16* __restrict__ Wt,
    const float* __restrict__ bias, float* __restrict__ C,
    int M, int do_relu)
{
  __shared__ float As[2][TM * TK];          // 2 x 16 KB
  const int tid   = threadIdx.x;
  const int lane  = tid & 31;
  const int wave  = tid >> 5;
  const int waveM = wave & 3;               // 4 row strips of 32
  const int waveN = wave >> 2;              // 2 col strips of 32
  const int row16 = lane & 15;
  const int hi    = lane >> 4;              // half-wave select
  const int blockM = blockIdx.x * TM;
  const int blockN = blockIdx.y * TN;

  // stage one 128x32 f32 A tile into LDS buffer b (4 async b128 per thread)
  auto stage = [&](int k0, int b) {
#pragma unroll
    for (int i = 0; i < 4; ++i) {
      int idx = tid + i * 256;
      int r = idx >> 3;
      int c = (idx & 7) << 2;
      int gr = blockM + r; if (gr >= M) gr = M - 1;    // clamp, masked at store
      unsigned lds  = (unsigned)(unsigned long long)&As[b][r * TK + c];
      unsigned goff = (unsigned)((gr * DD + k0 + c) * 4);
      async_load_b128(lds, A, goff);
    }
  };

  v8f acc[2][2] = {};
  stage(0, 0);

#pragma unroll
  for (int it = 0; it < 8; ++it) {
    const int k0 = it * TK;
    if (it < 7) {
      stage(k0 + TK, (it + 1) & 1);         // prefetch next tile (other buffer)
      // async loads complete in order: cnt<=4 => previous tile fully landed
      asm volatile("s_wait_asynccnt 0x4" ::: "memory");
    } else {
      asm volatile("s_wait_asynccnt 0x0" ::: "memory");
    }
    __syncthreads();

    const float* as = As[it & 1];

    // B fragments: col = lane&15 within 16-tile, K = k0 + hi*16 + e
    // pre-swizzled [n][k] bf16 -> 16 contiguous elements, 32B aligned
    const int bcol0 = blockN + waveN * 32 + row16;
    v16bf bf0 = *(const v16bf*)(Wt + (size_t)bcol0        * DD + k0 + hi * 16);
    v16bf bf1 = *(const v16bf*)(Wt + (size_t)(bcol0 + 16) * DD + k0 + hi * 16);

    // A fragments: row = lane&15 within 16-tile, K runs {kb..kb+15} (kb = hi*16)
    // Build from 2-wide vectors so every convert is one v_cvt_pk_bf16_f32.
    AFrag af[2];
#pragma unroll
    for (int t = 0; t < 2; ++t) {
      const int ar = waveM * 32 + t * 16 + row16;
      const int kb = hi * 16;
      const v2f* lo = (const v2f*)&as[ar * TK + kb];        // 8 consecutive f32
      const v2f* hi2 = (const v2f*)&as[ar * TK + kb + 8];   // 8 consecutive f32
#pragma unroll
      for (int e = 0; e < 4; ++e) af[t].h[e]     = __builtin_convertvector(lo[e],  v2bf);
#pragma unroll
      for (int e = 0; e < 4; ++e) af[t].h[4 + e] = __builtin_convertvector(hi2[e], v2bf);
    }

    acc[0][0] = __builtin_amdgcn_wmma_f32_16x16x32_bf16(false, af[0].v, false, bf0,
                                                        (short)0, acc[0][0], false, false);
    acc[0][1] = __builtin_amdgcn_wmma_f32_16x16x32_bf16(false, af[0].v, false, bf1,
                                                        (short)0, acc[0][1], false, false);
    acc[1][0] = __builtin_amdgcn_wmma_f32_16x16x32_bf16(false, af[1].v, false, bf0,
                                                        (short)0, acc[1][0], false, false);
    acc[1][1] = __builtin_amdgcn_wmma_f32_16x16x32_bf16(false, af[1].v, false, bf1,
                                                        (short)0, acc[1][1], false, false);
    __syncthreads();                         // compute done before buffer reuse
  }

  // store (C/D layout: VGPR r = M row, lane half selects +8)
#pragma unroll
  for (int tm = 0; tm < 2; ++tm)
#pragma unroll
    for (int tn = 0; tn < 2; ++tn)
#pragma unroll
      for (int r = 0; r < 8; ++r) {
        int row = blockM + waveM * 32 + tm * 16 + hi * 8 + r;
        int col = blockN + waveN * 32 + tn * 16 + row16;
        if (row < M) {
          float v = acc[tm][tn][r] + bias[col];
          if (do_relu) v = fmaxf(v, 0.f);
          C[row * DD + col] = v;
        }
      }
}

// ---------------------------------------------------------------------------
// small elementwise / fill kernels
// ---------------------------------------------------------------------------
__global__ void k_fill(float* __restrict__ p, int n, float v) {
  int t = blockIdx.x * blockDim.x + threadIdx.x;
  if (t < n) p[t] = v;
}

__global__ void k_add_pe(float* __restrict__ h, int n) {
  int t = blockIdx.x * blockDim.x + threadIdx.x;
  if (t >= n * DD) return;
  int row = t >> 8, d = t & 255;
  float pos  = (float)(row % PP);                 // batch = repeat(arange(B), P)
  int   j    = d >> 1;
  float ang  = pos * __expf(-(float)j * 0.0719558252f);  // ln(10000)/128
  h[t] += (d & 1) ? __cosf(ang) : __sinf(ang);
}

__global__ void k_rowbias(float* __restrict__ acc, int n, const float* __restrict__ b) {
  int t = blockIdx.x * blockDim.x + threadIdx.x;
  if (t < n * DD) acc[t] += b[t & 255];
}

__global__ void k_relu_scale(float* __restrict__ h, const float* __restrict__ acc,
                             int n, float inv) {
  int t = blockIdx.x * blockDim.x + threadIdx.x;
  if (t < n * DD) h[t] = fmaxf(acc[t] * inv, 0.f);
}

// ---------------------------------------------------------------------------
// GATv2 edge kernels: 3-pass segment softmax with L2-resident atomics
// ---------------------------------------------------------------------------
__global__ void k_edge_score(const int* __restrict__ ei, int E,
                             const float* __restrict__ xl, const float* __restrict__ xr,
                             const float* __restrict__ att,
                             float* __restrict__ score, float* __restrict__ mmax) {
  int t = blockIdx.x * blockDim.x + threadIdx.x;
  if (t >= E * HH) return;
  int e = t >> 2, h = t & 3;
  int src = ei[e], dst = ei[E + e];
  const float* pl = xl + src * DD + h * DH_;
  const float* pr = xr + dst * DD + h * DH_;
  const float* pa = att + h * DH_;
  float s = 0.f;
#pragma unroll 8
  for (int d = 0; d < DH_; ++d) {
    float v = pl[d] + pr[d];
    v = (v > 0.f) ? v : 0.2f * v;          // leaky_relu(0.2)
    s += v * pa[d];
  }
  score[t] = s;
  __hip_atomic_fetch_max(&mmax[dst * HH + h], s,
                         __ATOMIC_RELAXED, __HIP_MEMORY_SCOPE_AGENT);
}

__global__ void k_edge_exp(const int* __restrict__ ei, int E,
                           float* __restrict__ score, const float* __restrict__ mmax,
                           float* __restrict__ den) {
  int t = blockIdx.x * blockDim.x + threadIdx.x;
  if (t >= E * HH) return;
  int e = t >> 2, h = t & 3;
  int dst = ei[E + e];
  float ex = __expf(score[t] - mmax[dst * HH + h]);
  score[t] = ex;
  atomicAdd(&den[dst * HH + h], ex);
}

__global__ void k_edge_msg(const int* __restrict__ ei, int E,
                           const float* __restrict__ score, const float* __restrict__ den,
                           const float* __restrict__ xl, float* __restrict__ accd) {
  int t = blockIdx.x * blockDim.x + threadIdx.x;
  if (t >= E * HH) return;
  int e = t >> 2, h = t & 3;
  int src = ei[e], dst = ei[E + e];
  float alpha = score[t] / (den[dst * HH + h] + 1e-16f);
  const float* pl = xl + src * DD + h * DH_;
  float* pd = accd + dst * DD + h * DH_;
#pragma unroll 8
  for (int d = 0; d < DH_; ++d) atomicAdd(&pd[d], alpha * pl[d]);
}

// ---------------------------------------------------------------------------
// scatter-mean pooling over the 3*P rows of each scene
// ---------------------------------------------------------------------------
__global__ void k_pool(const float* __restrict__ ht, const float* __restrict__ ha,
                       const float* __restrict__ hv, float* __restrict__ s) {
  int b = blockIdx.x, d = threadIdx.x;
  float acc = 0.f;
  int base = b * PP;
  for (int p = 0; p < PP; ++p) {
    int r = (base + p) * DD + d;
    acc += ht[r] + ha[r] + hv[r];
  }
  s[b * DD + d] = acc / (3.0f * PP);
}

__global__ void k_l2norm(float* __restrict__ s) {
  __shared__ float buf[DD];
  int b = blockIdx.x, d = threadIdx.x;
  float v = s[b * DD + d];
  buf[d] = v * v;
  __syncthreads();
  for (int st = 128; st > 0; st >>= 1) {
    if (d < st) buf[d] += buf[d + st];
    __syncthreads();
  }
  s[b * DD + d] = v / (sqrtf(buf[0]) + 1e-12f);
}

// ---------------------------------------------------------------------------
// InfoNCE loss, single block of 64 threads, two-pass logsumexp per row
// ---------------------------------------------------------------------------
static __device__ float contrast_row(const float* sa, const float* sb, int i) {
  const float invT = 2.0f;                 // 1 / TEMP
  float mx = -1e30f;
  for (int j = 0; j < 2 * NB; ++j) {
    const float* bptr = (j < NB) ? (sb + j * DD) : (sa + (j - NB) * DD);
    float dot = 0.f;
    for (int d = 0; d < DD; ++d) dot += sa[i * DD + d] * bptr[d];
    float lg = dot * invT;
    if (j >= NB && (j - NB) == i) lg -= 1e9f;
    mx = fmaxf(mx, lg);
  }
  float se = 0.f, diag = 0.f;
  for (int j = 0; j < 2 * NB; ++j) {
    const float* bptr = (j < NB) ? (sb + j * DD) : (sa + (j - NB) * DD);
    float dot = 0.f;
    for (int d = 0; d < DD; ++d) dot += sa[i * DD + d] * bptr[d];
    float lg = dot * invT;
    if (j >= NB && (j - NB) == i) lg -= 1e9f;
    if (j == i) diag = lg;
    se += __expf(lg - mx);
  }
  return mx + __logf(se) - diag;
}

__global__ void k_loss(const float* __restrict__ s1, const float* __restrict__ s2,
                       float* __restrict__ out) {
  __shared__ float red[NB];
  int i = threadIdx.x;
  red[i] = contrast_row(s1, s2, i) + contrast_row(s2, s1, i);
  __syncthreads();
  for (int st = 32; st > 0; st >>= 1) {
    if (i < st) red[i] += red[i + st];
    __syncthreads();
  }
  if (i == 0) out[0] = red[0] / (float)NB;
}

// ---------------------------------------------------------------------------
// host orchestration (all launches on `stream`; d_ws needs ~290 MB scratch)
// ---------------------------------------------------------------------------
extern "C" void kernel_launch(void* const* d_in, const int* in_sizes, int n_in,
                              void* d_out, int out_size, void* d_ws, size_t ws_size,
                              hipStream_t stream) {
  (void)in_sizes; (void)n_in; (void)out_size; (void)ws_size;

  const float* W_in = (const float*)d_in[8];
  const float* b_in = (const float*)d_in[9];
  const float* Wl   = (const float*)d_in[10];
  const float* bl   = (const float*)d_in[11];
  const float* Wr   = (const float*)d_in[12];
  const float* br   = (const float*)d_in[13];
  const float* att  = (const float*)d_in[14];
  const float* bo   = (const float*)d_in[15];
  const float* W_p0 = (const float*)d_in[16];
  const float* b_p0 = (const float*)d_in[17];
  const float* W_p1 = (const float*)d_in[18];
  const float* b_p1 = (const float*)d_in[19];
  const int* ei[RR];
  for (int k = 0; k < RR; ++k) ei[k] = (const int*)d_in[23 + k];

  // workspace carve-out (float units)
  float* ws = (float*)d_ws;
  size_t off = 0;
  auto carve = [&](size_t n) { float* p = ws + off; off += n; return p; };
  float* h[4];   for (int t = 0; t < 3; ++t) h[t]   = carve((size_t)NN * DD);
  h[3] = carve((size_t)NZ_ * DD);
  float* acc[4]; for (int t = 0; t < 3; ++t) acc[t] = carve((size_t)NN * DD);
  acc[3] = carve((size_t)NZ_ * DD);
  float* xl    = carve((size_t)NN * DD);
  float* xr    = carve((size_t)NN * DD);
  float* score = carve((size_t)ECROSS * HH);
  float* mmax  = carve((size_t)NN * HH);
  float* den   = carve((size_t)NN * HH);
  float* s1    = carve((size_t)NB * DD);
  float* s2    = carve((size_t)NB * DD);
  float* stmp  = carve((size_t)NB * DD);
  // bf16 transposed weights: 3 + 26 + 26 + 1 + 1 = 57 matrices of 64K elems
  __bf16* wt_in = (__bf16*)carve((size_t)57 * DD * DD / 2);
  __bf16* wt_l  = wt_in + (size_t)3  * DD * DD;
  __bf16* wt_r  = wt_l  + (size_t)26 * DD * DD;
  __bf16* wt_p0 = wt_r  + (size_t)26 * DD * DD;
  __bf16* wt_p1 = wt_p0 + (size_t)DD * DD;

  // one-time weight convert + transpose (bf16 [N,K] layout)
  auto wswz = [&](const float* src, __bf16* dst, int nmat) {
    int n = nmat * DD * DD;
    k_wswz<<<(n + 255) / 256, 256, 0, stream>>>(src, dst, nmat);
  };
  wswz(W_in, wt_in, 3);
  wswz(Wl,   wt_l,  26);
  wswz(Wr,   wt_r,  26);
  wswz(W_p0, wt_p0, 1);
  wswz(W_p1, wt_p1, 1);

  auto gemm = [&](const float* A, const __bf16* Wt, const float* bvec, float* C,
                  int M, int relu) {
    dim3 g((M + TM - 1) / TM, DD / TN);
    k_gemm256<<<g, 256, 0, stream>>>(A, Wt, bvec, C, M, relu);
  };
  auto fill = [&](float* p, int n, float v) {
    k_fill<<<(n + 255) / 256, 256, 0, stream>>>(p, n, v);
  };

  const int rel_s[RR] = {0,1,0,2,1,2,0,3,1,3,2,3,3};
  const int rel_d[RR] = {1,0,2,0,2,1,3,0,3,1,3,2,3};
  const int ecnt[RR]  = {ECROSS,ECROSS,ECROSS,ECROSS,ECROSS,ECROSS,
                         NN,NN,NN,NN,NN,NN,NZ_};
  const int nnod[4]   = {NN, NN, NN, NZ_};
  const float dstc[4] = {3.f, 3.f, 3.f, 4.f};

  for (int view = 0; view < 2; ++view) {
    const float* xin[4] = {(const float*)d_in[view * 4 + 0],
                           (const float*)d_in[view * 4 + 1],
                           (const float*)d_in[view * 4 + 2],
                           (const float*)d_in[view * 4 + 3]};
    float* sOut = view ? s2 : s1;

    // input projections (z reuses the text linear) + sinusoidal PE
    for (int m = 0; m < 3; ++m)
      gemm(xin[m], wt_in + (size_t)m * DD * DD, b_in + m * DD, h[m], NN, 0);
    gemm(xin[3], wt_in, b_in, h[3], NZ_, 0);
    for (int m = 0; m < 3; ++m)
      k_add_pe<<<(NN * DD + 255) / 256, 256, 0, stream>>>(h[m], NN);

    for (int l = 0; l < LL; ++l) {
      for (int t = 0; t < 4; ++t) fill(acc[t], nnod[t] * DD, 0.f);
      for (int k = 0; k < RR; ++k) {
        int st = rel_s[k], dt = rel_d[k];
        size_t midx = (size_t)(l * RR + k);
        size_t vofs = midx * DD;
        gemm(h[st], wt_l + midx * DD * DD, bl + vofs, xl, nnod[st], 0);
        gemm(h[dt], wt_r + midx * DD * DD, br + vofs, xr, nnod[dt], 0);
        fill(mmax, nnod[dt] * HH, -3e38f);
        fill(den,  nnod[dt] * HH, 0.f);
        int Ek = ecnt[k];
        int nt = Ek * HH, nb = (nt + 255) / 256;
        k_edge_score<<<nb, 256, 0, stream>>>(ei[k], Ek, xl, xr, att + vofs,
                                             score, mmax);
        k_edge_exp<<<nb, 256, 0, stream>>>(ei[k], Ek, score, mmax, den);
        k_edge_msg<<<nb, 256, 0, stream>>>(ei[k], Ek, score, den, xl, acc[dt]);
        k_rowbias<<<(nnod[dt] * DD + 255) / 256, 256, 0, stream>>>(
            acc[dt], nnod[dt], bo + vofs);
      }
      for (int t = 0; t < 4; ++t)
        k_relu_scale<<<(nnod[t] * DD + 255) / 256, 256, 0, stream>>>(
            h[t], acc[t], nnod[t], 1.0f / dstc[t]);
    }

    // pooling -> projection head (fused ReLU) -> L2 normalize
    k_pool<<<NB, DD, 0, stream>>>(h[0], h[1], h[2], sOut);
    gemm(sOut, wt_p0, b_p0, stmp, NB, 1);
    gemm(stmp, wt_p1, b_p1, sOut, NB, 1);
    k_l2norm<<<NB, DD, 0, stream>>>(sOut);
  }

  k_loss<<<1, NB, 0, stream>>>(s1, s2, (float*)d_out);
}
